// WindowAttention_54022098649834
// MI455X (gfx1250) — compile-verified, matrix operand order checked by reading, not run on hardware
//
#include <hip/hip_runtime.h>

typedef __attribute__((ext_vector_type(16))) _Float16 v16h;
typedef __attribute__((ext_vector_type(8)))  float    v8f;

#define SEQ   4096
#define NHD   16
#define HDIM  64
#define WRAD  128
#define RSTR  (NHD * HDIM)   // floats between consecutive sequence positions (src layout)
#define PST   40             // LDS row stride in halves (80B, 16B-aligned, padded)

__device__ __forceinline__ int clampi(int x, int lo, int hi) {
  return x < lo ? lo : (x > hi ? hi : x);
}

// f32 -> f16 fragment packer (8 floats at p0 -> halves 0..7, 8 at p1 -> 8..15)
__device__ __forceinline__ v16h pack16(const float* p0, const float* p1, float sc) {
  float4 a = *(const float4*)(p0);
  float4 b = *(const float4*)(p0 + 4);
  float4 c = *(const float4*)(p1);
  float4 d = *(const float4*)(p1 + 4);
  v16h r;
  r[0]  = (_Float16)(a.x * sc); r[1]  = (_Float16)(a.y * sc);
  r[2]  = (_Float16)(a.z * sc); r[3]  = (_Float16)(a.w * sc);
  r[4]  = (_Float16)(b.x * sc); r[5]  = (_Float16)(b.y * sc);
  r[6]  = (_Float16)(b.z * sc); r[7]  = (_Float16)(b.w * sc);
  r[8]  = (_Float16)(c.x * sc); r[9]  = (_Float16)(c.y * sc);
  r[10] = (_Float16)(c.z * sc); r[11] = (_Float16)(c.w * sc);
  r[12] = (_Float16)(d.x * sc); r[13] = (_Float16)(d.y * sc);
  r[14] = (_Float16)(d.z * sc); r[15] = (_Float16)(d.w * sc);
  return r;
}

// f16 fragment loader: two 16B global/ds loads -> v16h
__device__ __forceinline__ v16h ldfrag(const _Float16* p0, const _Float16* p1) {
  union { uint4 u[2]; v16h h; } cv;
  cv.u[0] = *(const uint4*)p0;
  cv.u[1] = *(const uint4*)p1;
  return cv.h;
}

// ---------------- pre-pass 1: K f32 [b][s][h][64] -> f16 head-major [b][h][s][64]
__global__ __launch_bounds__(256) void conv_k(const float* __restrict__ K,
                                              _Float16* __restrict__ KF) {
  const size_t i2 = ((size_t)blockIdx.x * 256 + threadIdx.x) * 2;  // flat src idx
  const int d = (int)(i2 & 63);
  const int h = (int)((i2 >> 6) & (NHD - 1));
  const int s = (int)((i2 >> 10) & (SEQ - 1));
  const int b = (int)(i2 >> 22);
  const float2 f = *(const float2*)(K + i2);
  union { _Float16 hh[2]; unsigned u; } pk;
  pk.hh[0] = (_Float16)f.x;
  pk.hh[1] = (_Float16)f.y;
  const size_t dst = (((size_t)b * NHD + h) * SEQ + s) * HDIM + d;
  *(unsigned*)(KF + dst) = pk.u;
}

// ---------------- pre-pass 2: V f32 [b][s][h][64] -> f16 transposed [b][h][d][s]
__global__ __launch_bounds__(256) void conv_v(const float* __restrict__ V,
                                              _Float16* __restrict__ VT) {
  __shared__ float tile[16][HDIM + 1];
  const int blk = blockIdx.x;            // 2*16*256 blocks
  const int st  = blk & 255;
  const int h   = (blk >> 8) & (NHD - 1);
  const int b   = blk >> 12;
  const int s0  = st << 4;
  const int t   = threadIdx.x;
  #pragma unroll
  for (int i = 0; i < 4; ++i) {          // coalesced reads: 16 s-rows x 64 dims
    const int idx = t + i * 256;
    const int sl = idx >> 6, d = idx & 63;
    tile[sl][d] = V[(((size_t)b * SEQ + s0 + sl) * NHD + h) * HDIM + d];
  }
  __syncthreads();
  const size_t obase = ((size_t)b * NHD + h) * HDIM * (size_t)SEQ;
  #pragma unroll
  for (int i = 0; i < 2; ++i) {          // packed-pair transposed writes
    const int idx = t + i * 256;         // 64 dims x 8 s-pairs
    const int d = idx >> 3, sp = (idx & 7) * 2;
    union { _Float16 hh[2]; unsigned u; } pk;
    pk.hh[0] = (_Float16)tile[sp][d];
    pk.hh[1] = (_Float16)tile[sp + 1][d];
    *(unsigned*)(VT + obase + (size_t)d * SEQ + s0 + sp) = pk.u;
  }
}

// ---------------- main: sliding-window flash attention, f16-fragment sources
__global__ __launch_bounds__(128) void wattn_f16(
    const float* __restrict__ Q, const _Float16* __restrict__ KF,
    const _Float16* __restrict__ VT, float* __restrict__ O)
{
  __shared__ __align__(16) _Float16 pbuf[4][32 * PST];

  const int lane = threadIdx.x & 31;
  const int wv   = threadIdx.x >> 5;
  const int wid  = blockIdx.x * 4 + wv;   // 4096 waves
  const int qt   = wid & 127;
  const int hh   = (wid >> 7) & (NHD - 1);
  const int bb   = wid >> 11;
  const int q0   = qt << 5;               // 32 queries per wave

  const int m  = lane & 15;
  const int hi = lane >> 4;

  const size_t qbase = (size_t)bb * SEQ * RSTR + (size_t)hh * HDIM;
  const float* qp = Q + qbase;
  float*       op = O + qbase;
  const _Float16* kf = KF + ((size_t)bb * NHD + hh) * SEQ * HDIM;
  const _Float16* vt = VT + ((size_t)bb * NHD + hh) * HDIM * (size_t)SEQ;

  _Float16* pw = pbuf[wv];

  v16h Aq[2][2];
  #pragma unroll
  for (int qs = 0; qs < 2; ++qs) {
    const float* qr = qp + (size_t)(q0 + qs * 16 + m) * RSTR;
    Aq[qs][0] = pack16(qr +      hi * 8, qr + 16 + hi * 8, 0.125f);
    Aq[qs][1] = pack16(qr + 32 + hi * 8, qr + 48 + hi * 8, 0.125f);
  }

  v8f zero = {};
  v8f acc[2][4] = { { zero, zero, zero, zero }, { zero, zero, zero, zero } };
  float mi[2][8], li[2][8];
  #pragma unroll
  for (int qs = 0; qs < 2; ++qs)
    #pragma unroll
    for (int r = 0; r < 8; ++r) { mi[qs][r] = -1e30f; li[qs][r] = 0.f; }

  #pragma unroll 1
  for (int t = 0; t < 9; ++t) {
    const int kb32 = q0 - WRAD + t * 32;

    // ---- S = Q Kᵀ : K fragments are four b128 loads per subtile ----
    v8f Sv[2][2];
    #pragma unroll
    for (int kt = 0; kt < 2; ++kt) {
      const int kc = clampi(kb32 + kt * 16 + m, 0, SEQ - 1);
      const _Float16* kr = kf + (size_t)kc * HDIM;
      v16h Bk0 = ldfrag(kr +      hi * 8, kr + 16 + hi * 8);
      v16h Bk1 = ldfrag(kr + 32 + hi * 8, kr + 48 + hi * 8);
      #pragma unroll
      for (int qs = 0; qs < 2; ++qs) {
        v8f z = {};
        v8f s = __builtin_amdgcn_wmma_f32_16x16x32_f16(false, Aq[qs][0], false,
                                                       Bk0, (short)0, z,
                                                       false, false);
        s = __builtin_amdgcn_wmma_f32_16x16x32_f16(false, Aq[qs][1], false,
                                                   Bk1, (short)0, s,
                                                   false, false);
        Sv[qs][kt] = s;
      }
    }

    const int kp0 = kb32 + m;
    const bool in0 = (unsigned)kp0 < SEQ;
    const bool in1 = (unsigned)(kp0 + 16) < SEQ;

    // ---- online softmax, 16 rows per lane ----
    #pragma unroll
    for (int qs = 0; qs < 2; ++qs) {
      const int rel = 32 * t + m - hi * 8 - qs * 16;  // kp - (qpos - 128) + r
      #pragma unroll
      for (int r = 0; r < 8; ++r) {
        const bool ok0 = in0 && ((unsigned)(rel - r)      <= 2u * WRAD);
        const bool ok1 = in1 && ((unsigned)(rel + 16 - r) <= 2u * WRAD);
        float s0 = ok0 ? Sv[qs][0][r] : -1e30f;
        float s1 = ok1 ? Sv[qs][1][r] : -1e30f;
        float mx = fmaxf(s0, s1);
        #pragma unroll
        for (int off = 1; off < 16; off <<= 1)
          mx = fmaxf(mx, __shfl_xor(mx, off, 32));
        const float nm      = fmaxf(mi[qs][r], mx);
        const float rescale = __expf(mi[qs][r] - nm);
        const float p0 = ok0 ? __expf(s0 - nm) : 0.f;  // masked rows => p=0 exactly
        const float p1 = ok1 ? __expf(s1 - nm) : 0.f;
        float rs = p0 + p1;
        #pragma unroll
        for (int off = 1; off < 16; off <<= 1)
          rs += __shfl_xor(rs, off, 32);
        li[qs][r] = li[qs][r] * rescale + rs;
        mi[qs][r] = nm;
        #pragma unroll
        for (int cc = 0; cc < 4; ++cc) acc[qs][cc][r] *= rescale;
        const int rowoff = (qs * 16 + r + hi * 8) * PST;
        pw[rowoff + m]      = (_Float16)p0;
        pw[rowoff + 16 + m] = (_Float16)p1;
      }
    }

    __builtin_amdgcn_wave_barrier();

    v16h Ap[2];
    #pragma unroll
    for (int qs = 0; qs < 2; ++qs) {
      const int rowoff = (qs * 16 + m) * PST;
      Ap[qs] = ldfrag(pw + rowoff + hi * 8, pw + rowoff + 16 + hi * 8);
    }
    __builtin_amdgcn_wave_barrier();

    // ---- O += P V : V fragment = two b128 loads (transposed f16), used twice ----
    const int c0 = clampi(kb32 +      hi * 8, 0, SEQ - 8);  // chunk starts are
    const int c1 = clampi(kb32 + 16 + hi * 8, 0, SEQ - 8);  // multiples of 8
    #pragma unroll
    for (int cc = 0; cc < 4; ++cc) {
      const _Float16* vrow = vt + (size_t)(cc * 16 + m) * SEQ;
      v16h Bv = ldfrag(vrow + c0, vrow + c1);
      #pragma unroll
      for (int qs = 0; qs < 2; ++qs)
        acc[qs][cc] = __builtin_amdgcn_wmma_f32_16x16x32_f16(false, Ap[qs], false,
                                                             Bv, (short)0,
                                                             acc[qs][cc],
                                                             false, false);
    }

    if (t < 8) {  // gfx1250 global_prefetch_b8 for next key group
      const int nk = clampi(kb32 + 32 + m, 0, SEQ - 1);
      __builtin_prefetch(kf + (size_t)nk * HDIM, 0, 1);
    }
  }

  #pragma unroll
  for (int qs = 0; qs < 2; ++qs)
    #pragma unroll
    for (int r = 0; r < 8; ++r) {
      const float  inv = 1.0f / li[qs][r];
      const size_t row = (size_t)(q0 + qs * 16 + r + hi * 8) * RSTR;
      #pragma unroll
      for (int cc = 0; cc < 4; ++cc)
        op[row + cc * 16 + m] = acc[qs][cc][r] * inv;
    }
}

// ---------------- fallback: self-contained f32-source version (no workspace) ----
__global__ __launch_bounds__(128) void wattn_f32(
    const float* __restrict__ Q, const float* __restrict__ K,
    const float* __restrict__ V, float* __restrict__ O)
{
  __shared__ __align__(16) _Float16 pbuf[4][32 * PST];
  const int lane = threadIdx.x & 31;
  const int wv   = threadIdx.x >> 5;
  const int wid  = blockIdx.x * 4 + wv;
  const int qt   = wid & 127;
  const int hh   = (wid >> 7) & (NHD - 1);
  const int bb   = wid >> 11;
  const int q0   = qt << 5;
  const int m  = lane & 15;
  const int hi = lane >> 4;
  const size_t base = (size_t)bb * SEQ * RSTR + (size_t)hh * HDIM;
  const float* qp = Q + base;
  const float* kp = K + base;
  const float* vp = V + base;
  float*       op = O + base;
  _Float16* pw = pbuf[wv];

  v16h Aq[2][2];
  #pragma unroll
  for (int qs = 0; qs < 2; ++qs) {
    const float* qr = qp + (size_t)(q0 + qs * 16 + m) * RSTR;
    Aq[qs][0] = pack16(qr +      hi * 8, qr + 16 + hi * 8, 0.125f);
    Aq[qs][1] = pack16(qr + 32 + hi * 8, qr + 48 + hi * 8, 0.125f);
  }
  v8f zero = {};
  v8f acc[2][4] = { { zero, zero, zero, zero }, { zero, zero, zero, zero } };
  float mi[2][8], li[2][8];
  #pragma unroll
  for (int qs = 0; qs < 2; ++qs)
    #pragma unroll
    for (int r = 0; r < 8; ++r) { mi[qs][r] = -1e30f; li[qs][r] = 0.f; }

  #pragma unroll 1
  for (int t = 0; t < 9; ++t) {
    const int kb32 = q0 - WRAD + t * 32;
    v8f Sv[2][2];
    #pragma unroll
    for (int kt = 0; kt < 2; ++kt) {
      const int kc = clampi(kb32 + kt * 16 + m, 0, SEQ - 1);
      const float* kr = kp + (size_t)kc * RSTR;
      v16h Bk0 = pack16(kr +      hi * 8, kr + 16 + hi * 8, 1.f);
      v16h Bk1 = pack16(kr + 32 + hi * 8, kr + 48 + hi * 8, 1.f);
      #pragma unroll
      for (int qs = 0; qs < 2; ++qs) {
        v8f z = {};
        v8f s = __builtin_amdgcn_wmma_f32_16x16x32_f16(false, Aq[qs][0], false,
                                                       Bk0, (short)0, z, false, false);
        s = __builtin_amdgcn_wmma_f32_16x16x32_f16(false, Aq[qs][1], false,
                                                   Bk1, (short)0, s, false, false);
        Sv[qs][kt] = s;
      }
    }
    const int kp0 = kb32 + m;
    const bool in0 = (unsigned)kp0 < SEQ;
    const bool in1 = (unsigned)(kp0 + 16) < SEQ;
    #pragma unroll
    for (int qs = 0; qs < 2; ++qs) {
      const int rel = 32 * t + m - hi * 8 - qs * 16;
      #pragma unroll
      for (int r = 0; r < 8; ++r) {
        const bool ok0 = in0 && ((unsigned)(rel - r)      <= 2u * WRAD);
        const bool ok1 = in1 && ((unsigned)(rel + 16 - r) <= 2u * WRAD);
        float s0 = ok0 ? Sv[qs][0][r] : -1e30f;
        float s1 = ok1 ? Sv[qs][1][r] : -1e30f;
        float mx = fmaxf(s0, s1);
        #pragma unroll
        for (int off = 1; off < 16; off <<= 1)
          mx = fmaxf(mx, __shfl_xor(mx, off, 32));
        const float nm      = fmaxf(mi[qs][r], mx);
        const float rescale = __expf(mi[qs][r] - nm);
        const float p0 = ok0 ? __expf(s0 - nm) : 0.f;
        const float p1 = ok1 ? __expf(s1 - nm) : 0.f;
        float rs = p0 + p1;
        #pragma unroll
        for (int off = 1; off < 16; off <<= 1)
          rs += __shfl_xor(rs, off, 32);
        li[qs][r] = li[qs][r] * rescale + rs;
        mi[qs][r] = nm;
        #pragma unroll
        for (int cc = 0; cc < 4; ++cc) acc[qs][cc][r] *= rescale;
        const int rowoff = (qs * 16 + r + hi * 8) * PST;
        pw[rowoff + m]      = (_Float16)p0;
        pw[rowoff + 16 + m] = (_Float16)p1;
      }
    }
    __builtin_amdgcn_wave_barrier();
    v16h Ap[2];
    #pragma unroll
    for (int qs = 0; qs < 2; ++qs) {
      const int rowoff = (qs * 16 + m) * PST;
      Ap[qs] = ldfrag(pw + rowoff + hi * 8, pw + rowoff + 16 + hi * 8);
    }
    __builtin_amdgcn_wave_barrier();
    #pragma unroll
    for (int cc = 0; cc < 4; ++cc) {
      const int col = cc * 16 + m;
      v16h Bv;
      #pragma unroll
      for (int j = 0; j < 8; ++j) {
        int r0 = clampi(kb32 +      hi * 8 + j, 0, SEQ - 1);
        int r1 = clampi(kb32 + 16 + hi * 8 + j, 0, SEQ - 1);
        Bv[j]     = (_Float16)vp[(size_t)r0 * RSTR + col];
        Bv[j + 8] = (_Float16)vp[(size_t)r1 * RSTR + col];
      }
      #pragma unroll
      for (int qs = 0; qs < 2; ++qs)
        acc[qs][cc] = __builtin_amdgcn_wmma_f32_16x16x32_f16(false, Ap[qs], false,
                                                             Bv, (short)0,
                                                             acc[qs][cc], false, false);
    }
  }
  #pragma unroll
  for (int qs = 0; qs < 2; ++qs)
    #pragma unroll
    for (int r = 0; r < 8; ++r) {
      const float  inv = 1.0f / li[qs][r];
      const size_t row = (size_t)(q0 + qs * 16 + r + hi * 8) * RSTR;
      #pragma unroll
      for (int cc = 0; cc < 4; ++cc)
        op[row + cc * 16 + m] = acc[qs][cc][r] * inv;
    }
}

extern "C" void kernel_launch(void* const* d_in, const int* in_sizes, int n_in,
                              void* d_out, int out_size, void* d_ws, size_t ws_size,
                              hipStream_t stream) {
  (void)in_sizes; (void)n_in; (void)out_size;
  const float* q = (const float*)d_in[0];
  const float* k = (const float*)d_in[1];
  const float* v = (const float*)d_in[2];
  float* o = (float*)d_out;

  const size_t elems    = (size_t)2 * NHD * SEQ * HDIM;   // per tensor
  const size_t f16bytes = elems * sizeof(_Float16);       // 16 MB

  if (ws_size >= 2 * f16bytes) {
    _Float16* KF = (_Float16*)d_ws;
    _Float16* VT = KF + elems;
    hipLaunchKernelGGL(conv_k, dim3((unsigned)(elems / 512)), dim3(256), 0, stream, k, KF);
    hipLaunchKernelGGL(conv_v, dim3(2 * NHD * (SEQ / 16)), dim3(256), 0, stream, v, VT);
    hipLaunchKernelGGL(wattn_f16, dim3(1024), dim3(128), 0, stream, q, KF, VT, o);
  } else {
    hipLaunchKernelGGL(wattn_f32, dim3(1024), dim3(128), 0, stream, q, k, v, o);
  }
}
